// MCGDecoder_17763984736899
// MI455X (gfx1250) — matrix operand order repaired and checked by reading, not compiled, not run on hardware
//
#include <hip/hip_runtime.h>

#define DEVFN __device__ __forceinline__

typedef __bf16 bf16_t;
typedef __attribute__((ext_vector_type(16))) bf16_t v16bf;
typedef __attribute__((ext_vector_type(8)))  float  v8f;

// ---------------- problem sizes ----------------
constexpr int Bsz = 64, Asz = 128, Tn = 6, Dd = 256, Hh = 512, NBk = 5, OUTn = 401;
constexpr int GA   = 16;          // agents per workgroup
constexpr int ROWS = GA * Tn;     // 96 s-rows per WG; t-major: row = t*16 + a_local
constexpr int NPAD = 416;         // OUT padded to multiple of 16

// ---------------- bf16 weight workspace layout (ushort elements) ----------------
constexpr int DSQ     = Dd * Dd;           // 65536
constexpr int WSET    = NBk * DSQ;         // 327680
constexpr int OFF_SW1 = 0;
constexpr int OFF_SW2 = OFF_SW1 + WSET;
constexpr int OFF_CW1 = OFF_SW2 + WSET;
constexpr int OFF_CW2 = OFF_CW1 + WSET;
constexpr int OFF_DW1 = OFF_CW2 + WSET;    // dW1t: [512][256]
constexpr int OFF_DW2 = OFF_DW1 + Hh * Dd; // dW2t: [416][512]
constexpr int WS_ELEMS = OFF_DW2 + NPAD * Hh;   // 1,654,784 ushorts (~3.3 MB)

// ---------------- LDS layout (bytes) ----------------
constexpr int SP = 260;  // f32 pitch (256-wide buffers), bank-conflict-free fragment reads
constexpr int HP = 520;  // ushort pitch (bf16 activations, up to 512 wide)
constexpr int RP = 420;  // f32 pitch for decoder result (416 wide)
constexpr int OFF_sS   = 0;                        // 96 x SP f32   (99840 B)
constexpr int OFF_cS   = OFF_sS + ROWS * SP * 4;   // 16 x SP f32
constexpr int OFF_cnS  = OFF_cS + GA * SP * 4;     // 16 x SP f32  (end 133120)
// resS (96 x RP f32 = 161280 B) aliases sS/cS/cnS, used only in decoder phase
constexpr int OFF_hb   = ROWS * RP * 4;            // 161280, bf16 activations 96 x HP
constexpr int LDS_BYTES = OFF_hb + ROWS * HP * 2;  // 261120 B  (< 320KB/WGP)

// ---------------- bf16 helpers (round-to-nearest-even) ----------------
DEVFN unsigned short bfbits(float f) {
    union { float f; unsigned u; } a; a.f = f;
    unsigned r = a.u + 0x7FFFu + ((a.u >> 16) & 1u);
    return (unsigned short)(r >> 16);
}
DEVFN bf16_t f2bf(float f) {
    union { unsigned short s; bf16_t b; } o; o.s = bfbits(f);
    return o.b;
}

// ---------------- WMMA fragment loaders ----------------
// A (16x32 bf16) from f32 LDS, row-major with pitch (f32 elems).
// Lane L: M = L&15, half = L>>4; elems 0..7 -> k = kt*32 + half*8 + j,
//                                elems 8..15 -> k = kt*32 + 16 + half*8 + j.
DEVFN v16bf load_a_f32(const float* s, int pitch, int rowbase, int kt, int lane) {
    const float* p = s + (rowbase + (lane & 15)) * pitch + kt * 32 + (lane >> 4) * 8;
    v16bf a;
#pragma unroll
    for (int j = 0; j < 8; ++j) { a[j] = f2bf(p[j]); a[j + 8] = f2bf(p[j + 16]); }
    return a;
}
// A (16x32 bf16) from bf16 (ushort) LDS.
DEVFN v16bf load_a_bf16(const unsigned short* h, int pitch, int rowbase, int kt, int lane) {
    const unsigned short* p = h + (rowbase + (lane & 15)) * pitch + kt * 32 + (lane >> 4) * 8;
    union { uint4 q[2]; v16bf v; } u;
    u.q[0] = *(const uint4*)(p);
    u.q[1] = *(const uint4*)(p + 16);
    return u.v;
}
// B (32x16 bf16) from global pre-transposed weights Wt[N][K] (ushort).
// Lane L: N = L&15, half = L>>4; elems 0..7 -> k = kt*32 + half*8 + j,
// elems 8..15 -> k = kt*32 + 16 + half*8 + j.
DEVFN v16bf load_b(const unsigned short* Wt, int K, int nt, int kt, int lane) {
    const unsigned short* p = Wt + (nt * 16 + (lane & 15)) * K + kt * 32 + (lane >> 4) * 8;
    union { uint4 q[2]; v16bf v; } u;
    u.q[0] = *(const uint4*)(p);
    u.q[1] = *(const uint4*)(p + 16);
    return u.v;
}

DEVFN v8f wmma_bf16(v16bf a, v16bf b, v8f c) {
    return __builtin_amdgcn_wmma_f32_16x16x32_bf16(false, a, false, b, (short)0, c,
                                                   false, false);
}

// ================= kernel 1: f32 -> bf16 + transpose all weights =================
__global__ void mcg_convert_w(const float* __restrict__ sW1, const float* __restrict__ sW2,
                              const float* __restrict__ cW1, const float* __restrict__ cW2,
                              const float* __restrict__ dW1, const float* __restrict__ dW2,
                              unsigned short* __restrict__ ws) {
    int idx = blockIdx.x * blockDim.x + threadIdx.x;
    if (idx >= WS_ELEMS) return;
    float v;
    if (idx < 4 * WSET) {
        int seg = idx / WSET;
        int r   = idx % WSET;
        int i   = r / DSQ;
        int e   = r % DSQ;
        int n   = e / Dd, k = e % Dd;      // out[i][n][k] = in[i][k][n]
        const float* src = (seg == 0) ? sW1 : (seg == 1) ? sW2 : (seg == 2) ? cW1 : cW2;
        v = src[(i * Dd + k) * Dd + n];
    } else if (idx < 4 * WSET + Hh * Dd) {
        int e = idx - 4 * WSET;
        int n = e / Dd, k = e % Dd;        // dW1t[n][k] = dW1[k][n], dW1 is [256][512]
        v = dW1[k * Hh + n];
    } else {
        int e = idx - 4 * WSET - Hh * Dd;
        int n = e / Hh, k = e % Hh;        // dW2t[n][k] = dW2[k][n], dW2 is [512][401]
        v = (n < OUTn) ? dW2[k * OUTn + n] : 0.0f;
    }
    ws[idx] = bfbits(v);
}

// ================= kernel 2: fused recurrence + decoder + postprocess =================
__global__ __launch_bounds__(256)
void mcg_fused(const float* __restrict__ fe,      // (B,A,D)
               const float* __restrict__ anchor,  // (T,D)
               const float* __restrict__ sb1, const float* __restrict__ sb2,
               const float* __restrict__ cb1, const float* __restrict__ cb2,
               const float* __restrict__ db1, const float* __restrict__ db2,
               const unsigned short* __restrict__ ws,
               float* __restrict__ out) {
    extern __shared__ char smem[];
    float*          sS  = (float*)(smem + OFF_sS);    // 96 x SP f32
    float*          cS  = (float*)(smem + OFF_cS);    // 16 x SP f32
    float*          cnS = (float*)(smem + OFF_cnS);   // 16 x SP f32
    float*          resS = (float*)(smem);            // 96 x RP f32 (decoder only)
    unsigned short* hb  = (unsigned short*)(smem + OFF_hb); // 96 x HP bf16

    const int tid  = threadIdx.x;
    const int wg   = blockIdx.x;          // 512 WGs, 16 agents each
    const int w    = tid >> 5;            // wave id 0..7
    const int lane = tid & 31;
    const int m16  = lane & 15;
    const int half = lane >> 4;

    // ---- init: s = anchor broadcast, c = final_embedding ----
    for (int e = tid; e < ROWS * Dd; e += 256) {
        int row = e >> 8, d = e & 255;            // row = t*16 + a
        sS[row * SP + d] = anchor[(row >> 4) * Dd + d];
    }
    for (int e = tid; e < GA * Dd; e += 256) {
        int a = e >> 8, d = e & 255;
        cS[a * SP + d] = fe[(size_t)(wg * GA + a) * Dd + d];
    }
    __syncthreads();

    // ---- NB recurrence blocks ----
    for (int i = 0; i < NBk; ++i) {
        const unsigned short* w_s1 = ws + OFF_SW1 + i * DSQ;
        const unsigned short* w_s2 = ws + OFF_SW2 + i * DSQ;
        const unsigned short* w_c1 = ws + OFF_CW1 + i * DSQ;
        const unsigned short* w_c2 = ws + OFF_CW2 + i * DSQ;

        // ---- c layer1: ch1 = relu(c @ cW1 + cb1) -> hb rows 0..15 (bf16) ----
        for (int nt = w; nt < 16; nt += 8) {
            v8f acc = {};
#pragma unroll
            for (int kt = 0; kt < 8; ++kt)
                acc = wmma_bf16(load_a_f32(cS, SP, 0, kt, lane),
                                load_b(w_c1, Dd, nt, kt, lane), acc);
            int   col = nt * 16 + m16;
            float bb  = cb1[i * Dd + col];
#pragma unroll
            for (int r = 0; r < 8; ++r) {
                float v = acc[r] + bb;  v = v > 0.0f ? v : 0.0f;
                hb[(half * 8 + r) * HP + col] = bfbits(v);
            }
        }
        __syncthreads();

        // ---- c layer2: c_new = ch1 @ cW2 + cb2 -> cnS (f32) ----
        for (int nt = w; nt < 16; nt += 8) {
            v8f acc = {};
#pragma unroll
            for (int kt = 0; kt < 8; ++kt)
                acc = wmma_bf16(load_a_bf16(hb, HP, 0, kt, lane),
                                load_b(w_c2, Dd, nt, kt, lane), acc);
            int   col = nt * 16 + m16;
            float bb  = cb2[i * Dd + col];
#pragma unroll
            for (int r = 0; r < 8; ++r)
                cnS[(half * 8 + r) * SP + col] = acc[r] + bb;
        }
        __syncthreads();

        // ---- s layer1: sh1 = relu(s @ sW1 + sb1) -> hb rows 0..95 (bf16) ----
        for (int nt = w; nt < 16; nt += 8) {
            int   col = nt * 16 + m16;
            float bb  = sb1[i * Dd + col];
            for (int t = 0; t < Tn; ++t) {
                v8f acc = {};
#pragma unroll
                for (int kt = 0; kt < 8; ++kt)
                    acc = wmma_bf16(load_a_f32(sS, SP, t * 16, kt, lane),
                                    load_b(w_s1, Dd, nt, kt, lane), acc);
#pragma unroll
                for (int r = 0; r < 8; ++r) {
                    float v = acc[r] + bb;  v = v > 0.0f ? v : 0.0f;
                    hb[(t * 16 + half * 8 + r) * HP + col] = bfbits(v);
                }
            }
        }
        __syncthreads();

        // ---- s layer2 fused: s_new = (sh1 @ sW2 + sb2) * c_new ;
        //      max over t in registers ; s,c state updates in place ----
        for (int nt = w; nt < 16; nt += 8) {
            int   col = nt * 16 + m16;
            float bb  = sb2[i * Dd + col];
            float cc[8], mx[8];
#pragma unroll
            for (int r = 0; r < 8; ++r) {
                cc[r] = cnS[(half * 8 + r) * SP + col];
                mx[r] = -3.4e38f;
            }
            for (int t = 0; t < Tn; ++t) {                   // M-tile == timestep
                v8f acc = {};
#pragma unroll
                for (int kt = 0; kt < 8; ++kt)
                    acc = wmma_bf16(load_a_bf16(hb, HP, t * 16, kt, lane),
                                    load_b(w_s2, Dd, nt, kt, lane), acc);
#pragma unroll
                for (int r = 0; r < 8; ++r) {
                    float sn = (acc[r] + bb) * cc[r];
                    mx[r]   = fmaxf(mx[r], sn);
                    int off = (t * 16 + half * 8 + r) * SP + col;
                    sS[off] = 0.5f * sS[off] + 0.5f * sn;
                }
            }
#pragma unroll
            for (int r = 0; r < 8; ++r) {
                int off = (half * 8 + r) * SP + col;
                cS[off] = 0.5f * cS[off] + 0.5f * mx[r];
            }
        }
        __syncthreads();
    }

    // ---- decoder layer1: h = relu(s @ dW1 + db1) -> hb (96 x 512 bf16) ----
    const unsigned short* w_d1 = ws + OFF_DW1;
    for (int nt = w; nt < 32; nt += 8) {
        int   col = nt * 16 + m16;
        float bb  = db1[col];
        for (int t = 0; t < Tn; ++t) {
            v8f acc = {};
#pragma unroll
            for (int kt = 0; kt < 8; ++kt)
                acc = wmma_bf16(load_a_f32(sS, SP, t * 16, kt, lane),
                                load_b(w_d1, Dd, nt, kt, lane), acc);
#pragma unroll
            for (int r = 0; r < 8; ++r) {
                float v = acc[r] + bb;  v = v > 0.0f ? v : 0.0f;
                hb[(t * 16 + half * 8 + r) * HP + col] = bfbits(v);
            }
        }
    }
    __syncthreads();

    // ---- decoder layer2: res = h @ dW2 + db2 -> resS (96 x 416 f32, aliases state) ----
    const unsigned short* w_d2 = ws + OFF_DW2;
    for (int nt = w; nt < 26; nt += 8) {
        int   col = nt * 16 + m16;
        float bb  = (col < OUTn) ? db2[col] : 0.0f;
        for (int t = 0; t < Tn; ++t) {
            v8f acc = {};
#pragma unroll
            for (int kt = 0; kt < 16; ++kt)
                acc = wmma_bf16(load_a_bf16(hb, HP, t * 16, kt, lane),
                                load_b(w_d2, Hh, nt, kt, lane), acc);
#pragma unroll
            for (int r = 0; r < 8; ++r)
                resS[(t * 16 + half * 8 + r) * RP + col] = acc[r] + bb;
        }
    }
    __syncthreads();

    // ---- postprocess: probas / coordinates / covariance matrices ----
    const int PCO = Bsz * Asz * Tn;                 // 49152  (coords offset)
    const int CVO = PCO + Bsz * Asz * Tn * 80 * 2;  // 7913472 (cov offset)

    for (int e = tid; e < ROWS; e += 256) {         // probas = res[...,400]
        int t = e >> 4, al = e & 15;
        out[(wg * GA + al) * Tn + t] = resS[e * RP + 400];
    }
    for (int item = tid; item < ROWS * 80; item += 256) {
        int row = item / 80, j = item % 80;
        int t = row >> 4, al = row & 15;
        int bt = (wg * GA + al) * Tn + t;
        const float* rr = resS + row * RP;
        // coordinates
        out[PCO + (bt * 80 + j) * 2 + 0] = rr[2 * j];
        out[PCO + (bt * 80 + j) * 2 + 1] = rr[2 * j + 1];
        // covariance
        float av = rr[160 + j], bv = rr[240 + j], cv = rr[320 + j];
        float ec = __expf(cv);
        float ch = coshf(bv) * ec, sh = sinhf(bv) * ec;
        int cb = CVO + (bt * 80 + j) * 4;
        out[cb + 0] = __expf(av)  * ch;
        out[cb + 1] = sh;
        out[cb + 2] = sh;
        out[cb + 3] = __expf(-av) * ch;
    }
}

// ================= host launcher =================
extern "C" void kernel_launch(void* const* d_in, const int* in_sizes, int n_in,
                              void* d_out, int out_size, void* d_ws, size_t ws_size,
                              hipStream_t stream) {
    const float* fe     = (const float*)d_in[0];
    const float* anchor = (const float*)d_in[1];
    const float* sW1 = (const float*)d_in[2];
    const float* sb1 = (const float*)d_in[3];
    const float* sW2 = (const float*)d_in[4];
    const float* sb2 = (const float*)d_in[5];
    const float* cW1 = (const float*)d_in[6];
    const float* cb1 = (const float*)d_in[7];
    const float* cW2 = (const float*)d_in[8];
    const float* cb2 = (const float*)d_in[9];
    const float* dW1 = (const float*)d_in[10];
    const float* db1 = (const float*)d_in[11];
    const float* dW2 = (const float*)d_in[12];
    const float* db2 = (const float*)d_in[13];
    unsigned short* ws = (unsigned short*)d_ws;
    float*          op = (float*)d_out;

    mcg_convert_w<<<(WS_ELEMS + 255) / 256, 256, 0, stream>>>(sW1, sW2, cW1, cW2, dW1, dW2, ws);

    dim3 grid((Bsz * Asz) / GA);   // 512 workgroups
    mcg_fused<<<grid, 256, LDS_BYTES, stream>>>(fe, anchor, sb1, sb2, cb1, cb2,
                                                db1, db2, ws, op);
}
